// convPbc_21973052686655
// MI455X (gfx1250) — matrix operand order repaired and checked by reading, not compiled, not run on hardware
//
#include <hip/hip_runtime.h>
#include <hip/hip_bf16.h>

// CDNA5 (gfx1250) wave32 WMMA implicit-GEMM conv, bf16 inputs / f32 accumulate.
typedef __attribute__((ext_vector_type(16))) __bf16 v16bf;
typedef __attribute__((ext_vector_type(8)))  float  v8f;

#define LX 32
#define LY 32
#define CIN 256
#define COUT 512
#define FIL 4
#define KTOT (CIN * FIL * FIL)   // 4096
#define KTILES (KTOT / 32)       // 128 WMMA K-tiles (2 channels each)
#define NTILES (COUT / 16)       // 32 N-tiles

// ---------------------------------------------------------------------------
// Pack W (512,256,4,4) f32 -> bf16 in WMMA B-fragment layout:
//   dst[((ntile*128 + ktile)*32 + lane)*16 + 2*r + e]
// B as packed here (contiguous k): lanes 0-15 hold col=lane, K=0..15 (VGPR r: K=2r,2r+1);
//                    lanes 16-31 hold col=lane-16, K=16..31.
// GEMM K index within a K-tile: k = sel*16 + tap, sel = channel-in-pair,
// tap = u*4 + v.
// ---------------------------------------------------------------------------
__global__ __launch_bounds__(256) void pack_w_kernel(const float* __restrict__ W,
                                                     __bf16* __restrict__ wp) {
    int t = blockIdx.x * blockDim.x + threadIdx.x;   // 32*128*32 = 131072
    int lane = t & 31;
    int kt   = (t >> 5) & (KTILES - 1);
    int nt   = t >> 12;
    int col  = lane & 15;
    int h    = lane >> 4;
    int o    = nt * 16 + col;
    __bf16* dst = wp + (((size_t)nt * KTILES + kt) * 32 + lane) * 16;
#pragma unroll
    for (int r = 0; r < 8; ++r) {
#pragma unroll
        for (int e = 0; e < 2; ++e) {
            int k   = 16 * h + 2 * r + e;            // K index within tile
            int kg  = kt * 32 + k;                   // global GEMM K
            int c   = kg >> 4;                       // input channel
            int tap = kg & 15;
            int u   = tap >> 2;
            int v   = tap & 3;
            dst[2 * r + e] =
                (__bf16)W[(((size_t)o * CIN + c) * FIL + u) * FIL + v];
        }
    }
}

// ---------------------------------------------------------------------------
// Main kernel. Block = 256 threads (8 waves). blockIdx = (p, n).
// Block computes ALL 512 output channels for 32 pixels of row p:
//   wave w: mtile = w&1 (pixel q0 = 16*mtile), ntl = w>>1 (0..3);
//   wave owns 8 N-tiles { ntl + 4*i, i=0..7 } -> 8 x v8f accumulators.
// K loop: chunks of 8 input channels staged to LDS (1 channel per wave,
// division-free, software-pipelined), then 4 K-steps x 8 WMMAs per wave.
// One A-fragment per K-step feeds 8 WMMAs (A-gather amortized 8x).
// ---------------------------------------------------------------------------
__global__ __launch_bounds__(256) void conv_pbc_wmma_kernel(
    const float* __restrict__ x, const __bf16* __restrict__ wp,
    const float* __restrict__ bias, float* __restrict__ out) {

    const int p   = blockIdx.x;          // output row 0..31
    const int n   = blockIdx.y;          // batch 0..127
    const int tid = threadIdx.x;
    const int lane = tid & 31;
    const int w    = tid >> 5;           // wave id = staged channel-in-chunk
    const int mtile = w & 1;
    const int ntl   = w >> 1;            // 0..3

    // Staged pad tile: 8 channels x 4 rows x 35 cols (row stride 36) = 2.3 KB.
    __shared__ __bf16 lds[8 * 4 * 36];

    const int row = lane & 15;           // M (pixel) / N (channel) sub-index
    const int h   = lane >> 4;
    const int q   = mtile * 16 + row;    // this lane's base output column

    v8f acc[8];
#pragma unroll
    for (int i = 0; i < 8; ++i) {
        float bv = bias[(ntl + 4 * i) * 16 + row];
#pragma unroll
        for (int r = 0; r < 8; ++r) acc[i][r] = bv;
    }

    const float* xb = x + (size_t)n * CIN * (LX * LY);
    const v16bf* wlane = (const v16bf*)wp + lane;

    // ---- software-pipelined staging: regs hold next chunk's x values ----
    float stage[8];                      // [u][jj] : 4 rows x 2 column slabs
    auto load_chunk = [&](int c0) {
        const float* xc = xb + (size_t)(c0 + w) * (LX * LY);
#pragma unroll
        for (int u = 0; u < 4; ++u) {
            const int i = p + u;         // pad row, up to 34
#pragma unroll
            for (int jj = 0; jj < 2; ++jj) {
                const int j = lane + 32 * jj;    // pad col
                float val = 0.0f;
                if (j < 35) {
                    const int f = i * 35 + j;    // flat pad index
                    val = (f < LX * LY) ? xc[f] : 0.0f;
                    if (i >= LX && j >= LY) val += xc[(i - LX) * LY + (j - LY)];
                }
                stage[u * 2 + jj] = val;
            }
        }
    };

    load_chunk(0);

    for (int c0 = 0; c0 < CIN; c0 += 8) {
        __syncthreads();                 // LDS safe to overwrite
#pragma unroll
        for (int u = 0; u < 4; ++u) {
#pragma unroll
            for (int jj = 0; jj < 2; ++jj) {
                const int j = lane + 32 * jj;
                if (j < 35)
                    lds[(w * 4 + u) * 36 + j] = (__bf16)stage[u * 2 + jj];
            }
        }
        __syncthreads();

        if (c0 + 8 < CIN) load_chunk(c0 + 8);   // issue next chunk's loads

        // --- 4 WMMA K-steps (2 input channels each), 8 N-tiles per step ---
#pragma unroll
        for (int kk = 0; kk < 4; ++kk) {
            const int ktile = (c0 >> 1) + kk;
            // A fragment (16x32 bf16): lane half h picks K 16h..16h+15
            v16bf a;
#pragma unroll
            for (int r = 0; r < 8; ++r) {
                const int kb = 16 * h + 2 * r;
#pragma unroll
                for (int e = 0; e < 2; ++e) {
                    const int k   = kb + e;
                    const int sel = k >> 4;      // channel within pair
                    const int tap = k & 15;
                    const int uu  = tap >> 2;
                    const int vv  = tap & 3;
                    a[2 * r + e] =
                        lds[((2 * kk + sel) * 4 + uu) * 36 + q + vv];
                }
            }
#pragma unroll
            for (int i = 0; i < 8; ++i) {
                // B fragment: pre-swizzled, 32B per lane (2x global_load_b128)
                v16bf bfrag =
                    wlane[((size_t)(ntl + 4 * i) * KTILES + ktile) * 32];
                acc[i] = __builtin_amdgcn_wmma_f32_16x16x32_bf16(
                    /*neg_a=*/false, a, /*neg_b=*/false, bfrag,
                    /*c_mod=*/(short)0, acc[i],
                    /*reuse_a=*/false, /*reuse_b=*/false);
            }
        }
    }

    // --- epilogue: lane holds channel o_i, pixels q0+8h..q0+8h+7 (contig) ---
    const int qbase = mtile * 16 + 8 * h;
#pragma unroll
    for (int i = 0; i < 8; ++i) {
        const int o = (ntl + 4 * i) * 16 + row;
        float* op = out + (((size_t)n * COUT + o) * LX + p) * LY + qbase;
        *(v8f*)op = acc[i];              // 32B-aligned: two b128 stores
    }
}

extern "C" void kernel_launch(void* const* d_in, const int* in_sizes, int n_in,
                              void* d_out, int out_size, void* d_ws, size_t ws_size,
                              hipStream_t stream) {
    const float* x = (const float*)d_in[0];   // (128,256,32,32)
    const float* W = (const float*)d_in[1];   // (512,256,4,4)
    const float* b = (const float*)d_in[2];   // (512,)
    float* out = (float*)d_out;               // (128,512,32,32)

    __bf16* wp = (__bf16*)d_ws;               // 32*128*32*16 bf16 = 4 MB

    // Pack weights into WMMA-B layout (bf16).
    pack_w_kernel<<<(NTILES * KTILES * 32) / 256, 256, 0, stream>>>(W, wp);

    // Implicit-GEMM conv: grid (p=32, n=128), 256 threads (8 waves),
    // each block produces the full 512-channel, 32-pixel output row.
    dim3 grid(LX, 128);
    conv_pbc_wmma_kernel<<<grid, 256, 0, stream>>>(x, wp, b, out);
}
